// UnitRNN_90125593739371
// MI455X (gfx1250) — matrix operand order repaired and checked by reading, not compiled, not run on hardware
//
#include <hip/hip_runtime.h>

// ---------------------------------------------------------------------------
// 2-layer GRU scan (batch 1, hidden 128, 32768 steps) on gfx1250.
// Latency-bound recurrence -> one persistent workgroup on one WGP:
//   * recurrent weights resident in VGPRs as bf16 in WMMA A-operand layout
//   * hidden state in LDS (f32 master + bf16 broadcast copy)
//   * matvec via v_wmma_f32_16x16x32_bf16 with broadcast-B (all columns = h)
//   * 24 waves (768 thr), each wave owns one 16-row tile of the 384 gate rows
//   * all B vectors prefetched into VGPRs before each WMMA chain; x[t]
//     scalar-prefetched at step top so SMEM latency hides under phase-A WMMAs
// ---------------------------------------------------------------------------

typedef __attribute__((ext_vector_type(16))) __bf16 v16bf;
typedef __attribute__((ext_vector_type(8)))  float  v8f;

#define ISZ 8
#define HSZ 128
#define GSZ 384
#define NTHREADS 768   // 24 waves (wave32) -> 24 tiles of 16 rows = 384 rows

__device__ __forceinline__ float fast_sigmoid(float x) {
    return 1.0f / (1.0f + __expf(-x));
}
__device__ __forceinline__ float fast_tanh(float x) {
    // tanh(x) = 1 - 2/(exp(2x)+1), uses v_exp_f32 (transcendental unit)
    return 1.0f - 2.0f / (__expf(2.0f * x) + 1.0f);
}

__global__ void __launch_bounds__(NTHREADS, 1)
gru2_persistent_kernel(const float* __restrict__ x,
                       const float* __restrict__ w_ih0,
                       const float* __restrict__ w_hh0,
                       const float* __restrict__ b_ih0,
                       const float* __restrict__ b_hh0,
                       const float* __restrict__ w_ih1,
                       const float* __restrict__ w_hh1,
                       const float* __restrict__ b_ih1,
                       const float* __restrict__ b_hh1,
                       float* __restrict__ out,
                       int T)
{
    // ---- LDS state ----
    __shared__ float wih0_l[GSZ * ISZ];                    // 12 KB (input GEMV is tiny: keep VALU)
    __shared__ float bih0_l[GSZ], bhh0_l[GSZ];
    __shared__ float bih1_l[GSZ], bhh1_l[GSZ];
    __shared__ float gh0buf[GSZ];                          // Whh0 @ h0
    __shared__ float gh1hbuf[GSZ];                         // Whh1 @ h1
    __shared__ float gi1buf[GSZ];                          // Wih1 @ h0_new
    __shared__ float h0f[HSZ], h1f[HSZ];                   // f32 master hidden state
    __shared__ __align__(32) __bf16 h0bf[HSZ];             // bf16 broadcast copies (B operand src)
    __shared__ __align__(32) __bf16 h1bf[HSZ];

    const int tid  = threadIdx.x;
    const int lane = tid & 31;
    const int wv   = tid >> 5;           // wave id 0..23
    const int R    = wv * 16;            // this wave's 16-row tile base in [0,384)
    const int hi   = (lane >> 4) & 1;    // lane half (per ISA 16-bit A/B layouts)
    const int mrow = R + (lane & 15);    // gate row this lane loads A data for

    // ---- one-time LDS fill ----
    for (int i = tid; i < GSZ * ISZ; i += NTHREADS) wih0_l[i] = w_ih0[i];
    for (int i = tid; i < GSZ; i += NTHREADS) {
        bih0_l[i] = b_ih0[i];  bhh0_l[i] = b_hh0[i];
        bih1_l[i] = b_ih1[i];  bhh1_l[i] = b_hh1[i];
    }
    if (tid < HSZ) {
        h0f[tid] = 0.0f;  h1f[tid] = 0.0f;
        h0bf[tid] = (__bf16)0.0f;  h1bf[tid] = (__bf16)0.0f;
    }

    // ---- one-time: recurrent weights -> VGPRs in 16-bit A-operand layout ----
    // A 16x32 bf16 layout (matches B): lane half `hi`, element j : K = 32c + 16*hi + j
    v16bf Ahh0[4], Ahh1[4], Aih1[4];
    #pragma unroll
    for (int c = 0; c < 4; ++c) {
        const int kb = 32 * c + 16 * hi;
        #pragma unroll
        for (int j = 0; j < 8; ++j) {
            Ahh0[c][j]     = (__bf16)w_hh0[mrow * HSZ + kb + j];
            Ahh0[c][j + 8] = (__bf16)w_hh0[mrow * HSZ + kb + 8 + j];
            Ahh1[c][j]     = (__bf16)w_hh1[mrow * HSZ + kb + j];
            Ahh1[c][j + 8] = (__bf16)w_hh1[mrow * HSZ + kb + 8 + j];
            Aih1[c][j]     = (__bf16)w_ih1[mrow * HSZ + kb + j];
            Aih1[c][j + 8] = (__bf16)w_ih1[mrow * HSZ + kb + 8 + j];
        }
    }
    __syncthreads();

    // B 32x16 bf16 layout: lane half `hi` holds K = 32c + 16*hi + (0..15),
    // i.e. 16 *contiguous* bf16 -> one aligned 32B LDS vector load per chunk.
    const int boff  = 16 * hi;
    const int wlane = lane & 15;         // lanes 0 and 16 own the D rows (all D cols equal)
    const int mb    = R + 8 * hi;        // D rows this lane holds: mb..mb+7

    for (int t = 0; t < T; ++t) {
        // Prefetch x[t] (uniform address -> s_load_b256); latency hides under
        // phase-A WMMAs instead of stalling phase B behind two barriers.
        float xi[ISZ];
        #pragma unroll
        for (int i = 0; i < ISZ; ++i) xi[i] = x[(size_t)t * ISZ + i];

        // -- Phase A: gh0 = Whh0 @ h0 ; gh1h = Whh1 @ h1 (both depend only on old state)
        // Issue all 16 ds_load_b128 first so LDS latency overlaps the WMMA chain.
        v16bf b0v[4], b1v[4];
        #pragma unroll
        for (int c = 0; c < 4; ++c) {
            b0v[c] = *(const v16bf*)(h0bf + 32 * c + boff);
            b1v[c] = *(const v16bf*)(h1bf + 32 * c + boff);
        }
        v8f acc0 = {}; v8f acc1 = {};
        #pragma unroll
        for (int c = 0; c < 4; ++c) {
            acc0 = __builtin_amdgcn_wmma_f32_16x16x32_bf16(false, Ahh0[c], false, b0v[c],
                                                           (short)0, acc0, false, false);
            acc1 = __builtin_amdgcn_wmma_f32_16x16x32_bf16(false, Ahh1[c], false, b1v[c],
                                                           (short)0, acc1, false, false);
        }
        if (wlane == 0) {
            #pragma unroll
            for (int r = 0; r < 8; ++r) {
                gh0buf[mb + r]  = acc0[r];
                gh1hbuf[mb + r] = acc1[r];
            }
        }
        __syncthreads();

        // -- Phase B: layer-0 GRU cell (128 threads, one h element each)
        if (tid < HSZ) {
            float gir = bih0_l[tid];
            float giz = bih0_l[128 + tid];
            float gin = bih0_l[256 + tid];
            #pragma unroll
            for (int i = 0; i < ISZ; ++i) {
                gir = __builtin_fmaf(wih0_l[tid * ISZ + i],         xi[i], gir);
                giz = __builtin_fmaf(wih0_l[(128 + tid) * ISZ + i], xi[i], giz);
                gin = __builtin_fmaf(wih0_l[(256 + tid) * ISZ + i], xi[i], gin);
            }
            float ghr = gh0buf[tid]       + bhh0_l[tid];
            float ghz = gh0buf[128 + tid] + bhh0_l[128 + tid];
            float ghn = gh0buf[256 + tid] + bhh0_l[256 + tid];

            float rg  = fast_sigmoid(gir + ghr);
            float zg  = fast_sigmoid(giz + ghz);
            float ng  = fast_tanh(gin + rg * ghn);
            float h0n = (1.0f - zg) * ng + zg * h0f[tid];
            h0f[tid]  = h0n;
            h0bf[tid] = (__bf16)h0n;
        }
        __syncthreads();

        // -- Phase C: gi1 = Wih1 @ h0_new (prefetch all B chunks, then chain)
        v16bf b2v[4];
        #pragma unroll
        for (int c = 0; c < 4; ++c) {
            b2v[c] = *(const v16bf*)(h0bf + 32 * c + boff);
        }
        v8f acc2 = {};
        #pragma unroll
        for (int c = 0; c < 4; ++c) {
            acc2 = __builtin_amdgcn_wmma_f32_16x16x32_bf16(false, Aih1[c], false, b2v[c],
                                                           (short)0, acc2, false, false);
        }
        if (wlane == 0) {
            #pragma unroll
            for (int r = 0; r < 8; ++r) gi1buf[mb + r] = acc2[r];
        }
        __syncthreads();

        // -- Phase D: layer-1 GRU cell
        if (tid < HSZ) {
            float ir  = gi1buf[tid]        + bih1_l[tid];
            float iz  = gi1buf[128 + tid]  + bih1_l[128 + tid];
            float inn = gi1buf[256 + tid]  + bih1_l[256 + tid];
            float hr  = gh1hbuf[tid]       + bhh1_l[tid];
            float hz  = gh1hbuf[128 + tid] + bhh1_l[128 + tid];
            float hn  = gh1hbuf[256 + tid] + bhh1_l[256 + tid];

            float rg  = fast_sigmoid(ir + hr);
            float zg  = fast_sigmoid(iz + hz);
            float ng  = fast_tanh(inn + rg * hn);
            float h1n = (1.0f - zg) * ng + zg * h1f[tid];
            h1f[tid]  = h1n;
            h1bf[tid] = (__bf16)h1n;
        }
        __syncthreads();
    }

    if (tid < HSZ) out[tid] = h1f[tid];
}

extern "C" void kernel_launch(void* const* d_in, const int* in_sizes, int n_in,
                              void* d_out, int out_size, void* d_ws, size_t ws_size,
                              hipStream_t stream) {
    const float* x     = (const float*)d_in[0];
    const float* w_ih0 = (const float*)d_in[1];
    const float* w_hh0 = (const float*)d_in[2];
    const float* b_ih0 = (const float*)d_in[3];
    const float* b_hh0 = (const float*)d_in[4];
    const float* w_ih1 = (const float*)d_in[5];
    const float* w_hh1 = (const float*)d_in[6];
    const float* b_ih1 = (const float*)d_in[7];
    const float* b_hh1 = (const float*)d_in[8];
    float* out = (float*)d_out;

    const int T = in_sizes[0] / ISZ;   // 32768 sequential steps

    gru2_persistent_kernel<<<1, NTHREADS, 0, stream>>>(
        x, w_ih0, w_hh0, b_ih0, b_hh0, w_ih1, w_hh1, b_ih1, b_hh1, out, T);
}
